// GraphAttentionLayer_10574209483278
// MI455X (gfx1250) — compile-verified, matrix operand order checked by reading, not run on hardware
//
#include <hip/hip_runtime.h>
#include <hip/hip_bf16.h>

typedef _Float16 v16h __attribute__((ext_vector_type(16)));
typedef _Float16 v8h  __attribute__((ext_vector_type(8)));
typedef float    v8f  __attribute__((ext_vector_type(8)));
typedef int      vi4  __attribute__((ext_vector_type(4)));

#define ALPHA  0.2f
#define PHI    0.94247778f       // 3.1415926 * 0.3
#define A_PARA 0.5f

__device__ __forceinline__ float lrelu(float v) { return v > 0.f ? v : ALPHA * v; }

// ---------------------------------------------------------------------------
// CDNA5 async global->LDS staging (ASYNCcnt-tracked), with compile-safe guards
// Probe-derived signatures:
//   b128: (int __vector(4)* src_global, <lds ptr>, imm offset, imm cpol)
//   b32 : (int*             src_global, <lds ptr>, imm offset, imm cpol)
// ---------------------------------------------------------------------------
#define LDS_AS __attribute__((address_space(3)))

#if defined(__AMDGCN__) && __has_builtin(__builtin_amdgcn_global_load_async_to_lds_b128)
#define HAVE_ASYNC_LDS 1
#endif

__device__ __forceinline__ void copy16_async(void* lds, const void* glob) {
#if defined(HAVE_ASYNC_LDS)
  __builtin_amdgcn_global_load_async_to_lds_b128((vi4*)(glob),
                                                 (LDS_AS vi4*)(lds), 0, 0);
#else
  *(int4*)lds = *(const int4*)glob;
#endif
}

__device__ __forceinline__ void copy4_async(void* lds, const void* glob) {
#if defined(HAVE_ASYNC_LDS)
  __builtin_amdgcn_global_load_async_to_lds_b32((int*)(glob),
                                                (LDS_AS int*)(lds), 0, 0);
#else
  *(int*)lds = *(const int*)glob;
#endif
}

__device__ __forceinline__ void async_wait0() {
#if defined(HAVE_ASYNC_LDS)
#if __has_builtin(__builtin_amdgcn_s_wait_asynccnt)
  __builtin_amdgcn_s_wait_asynccnt(0);
#else
  asm volatile("s_wait_asynccnt 0x0" ::: "memory");
#endif
#endif
}

// ---------------------------------------------------------------------------
// Kernel 0: W [512][128] f32 -> k-chunk-blocked f16: wtb[kc][c][k&31], kc=k>>5
// ---------------------------------------------------------------------------
__global__ __launch_bounds__(256)
void k_transposeW(const float* __restrict__ W, _Float16* __restrict__ wtb) {
  int gid = blockIdx.x * 256 + threadIdx.x;     // 0..65535
  int k = gid >> 7, c = gid & 127;
  wtb[(size_t)(k >> 5) * 4096 + c * 32 + (k & 31)] = (_Float16)W[gid];
}

// ---------------------------------------------------------------------------
// Kernel 1: Wh = h @ W via WMMA f16 (f32 accum). One wave per 16-row block.
// Writes Wh j-chunk-blocked f16 (whtb[j>>5][col][j&31]) and Wh1/Wh2 scores.
// ---------------------------------------------------------------------------
__global__ __launch_bounds__(32)
void k_gemm(const float* __restrict__ h, const _Float16* __restrict__ wtb,
            const float* __restrict__ avec, _Float16* __restrict__ whtb,
            float* __restrict__ wh1, float* __restrict__ wh2) {
  const int lane = threadIdx.x & 31;
  const int n    = lane & 15;
  const int hlf  = lane >> 4;
  const int rowbase = blockIdx.x * 16;
  const int run0 = 8 * hlf, run1 = 16 + 8 * hlf;

  v8f acc[8] = {};

  for (int kc = 0; kc < 16; ++kc) {
    const float* hp = h + (size_t)(rowbase + n) * 512 + kc * 32;
    float4 f0 = *(const float4*)(hp + run0);
    float4 f1 = *(const float4*)(hp + run0 + 4);
    float4 f2 = *(const float4*)(hp + run1);
    float4 f3 = *(const float4*)(hp + run1 + 4);
    v16h ap;
    ap[0] = (_Float16)f0.x; ap[1] = (_Float16)f0.y; ap[2] = (_Float16)f0.z; ap[3] = (_Float16)f0.w;
    ap[4] = (_Float16)f1.x; ap[5] = (_Float16)f1.y; ap[6] = (_Float16)f1.z; ap[7] = (_Float16)f1.w;
    ap[8] = (_Float16)f2.x; ap[9] = (_Float16)f2.y; ap[10] = (_Float16)f2.z; ap[11] = (_Float16)f2.w;
    ap[12] = (_Float16)f3.x; ap[13] = (_Float16)f3.y; ap[14] = (_Float16)f3.z; ap[15] = (_Float16)f3.w;

    const _Float16* wb = wtb + (size_t)kc * 4096;
#pragma unroll
    for (int t = 0; t < 8; ++t) {
      v8h lo = *(const v8h*)(wb + (16 * t + n) * 32 + run0);
      v8h hi = *(const v8h*)(wb + (16 * t + n) * 32 + run1);
      v16h bp = __builtin_shufflevector(lo, hi, 0,1,2,3,4,5,6,7,8,9,10,11,12,13,14,15);
      acc[t] = __builtin_amdgcn_wmma_f32_16x16x32_f16(
          false, ap, false, bp, (short)0, acc[t], false, false);
    }
  }

  // Wh -> j-chunk-blocked f16 (row = j dimension of phase-2 B matrix)
#pragma unroll
  for (int t = 0; t < 8; ++t)
#pragma unroll
    for (int r = 0; r < 8; ++r) {
      int row = rowbase + r + 8 * hlf;
      int col = 16 * t + n;
      whtb[(size_t)(row >> 5) * 4096 + col * 32 + (row & 31)] = (_Float16)acc[t][r];
    }

  // attention scores: Wh1 = Wh@a[:128], Wh2 = Wh@a[128:]
  float a1v[8], a2v[8];
#pragma unroll
  for (int t = 0; t < 8; ++t) { a1v[t] = avec[16 * t + n]; a2v[t] = avec[128 + 16 * t + n]; }
#pragma unroll
  for (int r = 0; r < 8; ++r) {
    float s1 = 0.f, s2 = 0.f;
#pragma unroll
    for (int t = 0; t < 8; ++t) { s1 += acc[t][r] * a1v[t]; s2 += acc[t][r] * a2v[t]; }
#pragma unroll
    for (int m = 1; m < 16; m <<= 1) { s1 += __shfl_xor(s1, m, 32); s2 += __shfl_xor(s2, m, 32); }
    if (n == 0) {
      int row = rowbase + r + 8 * hlf;
      wh1[row] = s1; wh2[row] = s2;
    }
  }
}

// ---------------------------------------------------------------------------
// Kernel 2: global max of Wh2 (softmax stabilizer bound)
// ---------------------------------------------------------------------------
__global__ __launch_bounds__(1024)
void k_maxwh2(const float* __restrict__ wh2, float* __restrict__ mx) {
  __shared__ float red[32];
  int tid = threadIdx.x;
  float m = -3.4e38f;
  for (int i = tid; i < 8192; i += 1024) m = fmaxf(m, wh2[i]);
#pragma unroll
  for (int s = 1; s < 32; s <<= 1) m = fmaxf(m, __shfl_xor(m, s, 32));
  if ((tid & 31) == 0) red[tid >> 5] = m;
  __syncthreads();
  if (tid < 32) {
    float v = red[tid];
#pragma unroll
    for (int s = 1; s < 32; s <<= 1) v = fmaxf(v, __shfl_xor(v, s, 32));
    if (tid == 0) mx[0] = v;
  }
}

// ---------------------------------------------------------------------------
// Kernel 3: fused masked-softmax attention + aggregation (flash-style).
// Grid (128, 4): x = 4 row-blocks of 16 per workgroup, y = column split.
// Fixed stabilizer M_i = lrelu(Wh1_i + maxWh2) makes partials additive.
// Double-buffered LDS staging via CDNA5 async global->LDS loads (ASYNCcnt).
// ---------------------------------------------------------------------------
__global__ __launch_bounds__(128)
void k_attn(const int* __restrict__ adj, const float* __restrict__ wh1,
            const float* __restrict__ wh2, const float* __restrict__ mxp,
            const _Float16* __restrict__ whtb,
            float* __restrict__ hpAccP, float* __restrict__ denomP) {
  __shared__ int      adjLds[2][64 * 44];    // stride 44 ints: conflict-spread, 16B rows
  __shared__ _Float16 bLds[2][128 * 40];     // stride 40 f16: conflict-free, 16B rows
  __shared__ float    wh2Lds[2][32];

  const int tid  = threadIdx.x;
  const int w    = tid >> 5, lane = tid & 31;
  const int n    = lane & 15, hlf = lane >> 4;
  const int rowbase = blockIdx.x * 64 + w * 16;
  const int split   = blockIdx.y;
  const int run0 = 8 * hlf, run1 = 16 + 8 * hlf;
  const int NC = 64;

  const float wh1v = wh1[rowbase + n];
  const float mrow = lrelu(wh1v + mxp[0]);

  v8f acc[8] = {};
  float psum = 0.f;

  // stage one 32-column chunk into LDS buffer `buf` (all 128 threads)
  auto stageChunk = [&](int ci, int buf) {
    const int jbase = split * 2048 + ci * 32;
    {  // Wh B-chunk: contiguous 8 KB, 64B per thread
      const _Float16* src = whtb + (size_t)(jbase >> 5) * 4096 + tid * 32;
      _Float16* dst = &bLds[buf][tid * 40];
      copy16_async(dst + 0,  src + 0);
      copy16_async(dst + 8,  src + 8);
      copy16_async(dst + 16, src + 16);
      copy16_async(dst + 24, src + 24);
    }
    {  // adj tile 64x32: row-contiguous 128B segments, 64B per thread
      const int r = tid >> 1, hi16 = (tid & 1) * 16;
      const int* src = adj + (size_t)(blockIdx.x * 64 + r) * 8192 + jbase + hi16;
      int* dst = &adjLds[buf][r * 44 + hi16];
      copy16_async(dst + 0,  src + 0);
      copy16_async(dst + 4,  src + 4);
      copy16_async(dst + 8,  src + 8);
      copy16_async(dst + 12, src + 12);
    }
    if (tid < 32) copy4_async(&wh2Lds[buf][tid], wh2 + jbase + tid);
  };

  stageChunk(0, 0);

  for (int ci = 0; ci < NC; ++ci) {
    async_wait0();        // my wave's staged loads complete
    __syncthreads();      // everyone's staged loads complete
    if (ci + 1 < NC) stageChunk(ci + 1, (ci + 1) & 1);  // prefetch next chunk
    const int buf = ci & 1;

    // build probability A-tile in native 16x32 f16 WMMA layout
    const int arow = (w * 16 + n) * 44;
    float pv[16];
#pragma unroll
    for (int e = 0; e < 8; ++e) {
      int k = run0 + e;
      float ev = lrelu(wh1v + wh2Lds[buf][k]);
      pv[e] = adjLds[buf][arow + k] > 0 ? __expf(ev - mrow) : 0.f;
    }
#pragma unroll
    for (int e = 0; e < 8; ++e) {
      int k = run1 + e;
      float ev = lrelu(wh1v + wh2Lds[buf][k]);
      pv[8 + e] = adjLds[buf][arow + k] > 0 ? __expf(ev - mrow) : 0.f;
    }
    v16h ap;
#pragma unroll
    for (int e = 0; e < 16; ++e) { psum += pv[e]; ap[e] = (_Float16)pv[e]; }

#pragma unroll
    for (int t = 0; t < 8; ++t) {
      v8h lo = *(const v8h*)&bLds[buf][(16 * t + n) * 40 + run0];
      v8h hi = *(const v8h*)&bLds[buf][(16 * t + n) * 40 + run1];
      v16h bp = __builtin_shufflevector(lo, hi, 0,1,2,3,4,5,6,7,8,9,10,11,12,13,14,15);
      acc[t] = __builtin_amdgcn_wmma_f32_16x16x32_f16(
          false, ap, false, bp, (short)0, acc[t], false, false);
    }
  }

  float tot = psum + __shfl_xor(psum, 16, 32);
  if (lane < 16) denomP[(size_t)split * 8192 + rowbase + n] = tot;

  float* outp = hpAccP + (size_t)split * 8192 * 128;
#pragma unroll
  for (int t = 0; t < 8; ++t)
#pragma unroll
    for (int r = 0; r < 8; ++r)
      outp[(size_t)(rowbase + r + 8 * hlf) * 128 + 16 * t + n] = acc[t][r];
}

// ---------------------------------------------------------------------------
// Kernel 4: combine splits, normalize, project + logmap0, elu + 0.5*cos
// One wave per row (256-thread block = 8 rows).
// ---------------------------------------------------------------------------
__global__ __launch_bounds__(256)
void k_finalize(const float* __restrict__ hpAccP, const float* __restrict__ denomP,
                float* __restrict__ out) {
  const int w = threadIdx.x >> 5, lane = threadIdx.x & 31;
  const int row = blockIdx.x * 8 + w;

  float d = 0.f;
#pragma unroll
  for (int p = 0; p < 4; ++p) d += denomP[(size_t)p * 8192 + row];
  d = fmaxf(d, 1e-30f);

  float x[4];
#pragma unroll
  for (int q = 0; q < 4; ++q) {
    int c = lane + 32 * q;
    float s = 0.f;
#pragma unroll
    for (int p = 0; p < 4; ++p) s += hpAccP[(size_t)p * 8192 * 128 + (size_t)row * 128 + c];
    x[q] = s / d;
  }

  float ss = 0.f;
#pragma unroll
  for (int q = 0; q < 4; ++q) ss += x[q] * x[q];
#pragma unroll
  for (int m = 1; m < 32; m <<= 1) ss += __shfl_xor(ss, m, 32);

  float norm = fmaxf(sqrtf(ss), 1e-15f);
  const float maxnorm = 0.996f;                 // (1 - 4e-3)/sqrt(c), c=1
  float scale = norm > maxnorm ? maxnorm / norm : 1.0f;
  float pn = fmaxf(fminf(norm, maxnorm), 1e-15f);
  float z = fminf(pn, 1.f - 1e-7f);
  float at = 0.5f * __logf((1.f + z) / (1.f - z)); // artanh
  float fac = (at / pn) * scale;

#pragma unroll
  for (int q = 0; q < 4; ++q) {
    int c = lane + 32 * q;
    float y = x[q] * fac;
    float e = y > 0.f ? y : (__expf(y) - 1.f);   // elu
    out[(size_t)row * 128 + c] = e + A_PARA * __cosf(y + PHI);
  }
}

// ---------------------------------------------------------------------------
// Workspace layout (bytes):
//   [0,16MB)        hpAccP : 4 splits x 8192 x 128 f32
//   [16MB,18MB)     whtb   : 8192x128 f16 (j-chunk-blocked)
//   [18MB,+128KB)   wtb    : 512x128 f16 (k-chunk-blocked)
//   then wh1 (32KB), wh2 (32KB), denomP (128KB), maxWh2 (4B). Total ~19.2 MB.
// ---------------------------------------------------------------------------
extern "C" void kernel_launch(void* const* d_in, const int* in_sizes, int n_in,
                              void* d_out, int out_size, void* d_ws, size_t ws_size,
                              hipStream_t stream) {
  const float* h   = (const float*)d_in[0];   // [8192,512]
  const int*   adj = (const int*)d_in[1];     // [8192,8192]
  const float* W   = (const float*)d_in[2];   // [512,128]
  const float* a   = (const float*)d_in[3];   // [256,1]
  float* out = (float*)d_out;                 // [8192,128]

  char* ws = (char*)d_ws;
  float*    hpAccP = (float*)(ws + 0);
  _Float16* whtb   = (_Float16*)(ws + 16777216);
  _Float16* wtb    = (_Float16*)(ws + 18874368);
  float*    wh1    = (float*)(ws + 19005440);
  float*    wh2    = (float*)(ws + 19038208);
  float*    denomP = (float*)(ws + 19070976);
  float*    mx     = (float*)(ws + 19202048);

  k_transposeW<<<256, 256, 0, stream>>>(W, wtb);
  k_gemm<<<512, 32, 0, stream>>>(h, wtb, a, whtb, wh1, wh2);
  k_maxwh2<<<1, 1024, 0, stream>>>(wh2, mx);
  k_attn<<<dim3(128, 4), 128, 0, stream>>>(adj, wh1, wh2, mx, whtb, hpAccP, denomP);
  k_finalize<<<1024, 256, 0, stream>>>(hpAccP, denomP, out);
}
